// MyGCNNet_17386027614853
// MI455X (gfx1250) — matrix operand order repaired and checked by reading, not compiled, not run on hardware
//
#include <hip/hip_runtime.h>
#include <cstddef>

typedef __bf16 bf16_t;
typedef bf16_t v16bf __attribute__((ext_vector_type(16)));
typedef float  v8f   __attribute__((ext_vector_type(8)));

static constexpr int kB   = 32;
static constexpr int kHW  = 48 * 48;       // 2304
static constexpr int kN1  = kB * kHW;      // 73728 pixel nodes
static constexpr int kE1  = 8 * kN1;       // 589824 pixel edges
static constexpr int kSP  = 144;
static constexpr int kG   = kB * kSP;      // 4608 superpixel nodes
static constexpr int kE2  = 16 * kG;       // 73728 graph edges

__device__ __forceinline__ void atomAddF(float* p, float v) {
  // Lowers to global_atomic_add_f32 (no CAS loop) at agent scope.
  __hip_atomic_fetch_add(p, v, __ATOMIC_RELAXED, __HIP_MEMORY_SCOPE_AGENT);
}

// ---------------------------------------------------------------------------
// WMMA GEMM: out[M,128] = A[M,K](f32, cvt->bf16) @ Bw[K,128](f32, cvt->bf16) (+bias)
// 256 threads = 8 waves; each wave owns a 16-row tile x all 8 column tiles.
// Fragment layouts per CDNA5 ISA 7.12.2 (wave32).
// ---------------------------------------------------------------------------
template <int K>
__global__ void __launch_bounds__(256)
k_gemm128(const float* __restrict__ A, const float* __restrict__ Bw,
          const float* __restrict__ bias, float* __restrict__ out, int M) {
  constexpr int N = 128;
  __shared__ float Bs[K * N];
  const int tid = threadIdx.x;
  for (int i = tid; i < K * N; i += 256) Bs[i] = Bw[i];
  __syncthreads();

  const int wave = tid >> 5;
  const int lane = tid & 31;
  const int m0   = blockIdx.x * 128 + wave * 16;
  if (m0 >= M) return;
  const int lhi  = (lane & 16) ? 1 : 0;
  const int mrow = m0 + (lane & 15);
  const int ncol = lane & 15;

  v8f acc[8];
#pragma unroll
  for (int t = 0; t < 8; ++t)
#pragma unroll
    for (int j = 0; j < 8; ++j) acc[t][j] = 0.0f;

  // A 16x32 bf16 frag: lane<16 -> K {0..7,16..23}; lane>=16 -> +8.
  const float* arow = A + (size_t)mrow * K + (lhi ? 8 : 0);
#pragma unroll
  for (int kt = 0; kt < K / 32; ++kt) {
    const float* ar = arow + kt * 32;
    v16bf a;
#pragma unroll
    for (int e = 0; e < 8; ++e)  a[e] = (bf16_t)ar[e];
#pragma unroll
    for (int e = 8; e < 16; ++e) a[e] = (bf16_t)ar[e + 8];
    const int kb = kt * 32 + lhi * 8;  // B 32x16: lanes 0-15 K {0..7,16..23}; lanes 16-31 +8
#pragma unroll
    for (int nt = 0; nt < 8; ++nt) {
      v16bf b;
#pragma unroll
      for (int e = 0; e < 16; ++e) b[e] = (bf16_t)Bs[(kb + e + (e & 8)) * N + nt * 16 + ncol];
      acc[nt] = __builtin_amdgcn_wmma_f32_16x16x32_bf16(
          false, a, false, b, (short)0, acc[nt], false, false);
    }
  }

  // C/D: VGPR r -> row r + 8*lhi, col = ncol (per 16x16 f32 layout)
#pragma unroll
  for (int nt = 0; nt < 8; ++nt) {
#pragma unroll
    for (int r = 0; r < 8; ++r) {
      const int mm = m0 + r + lhi * 8;
      const int nn = nt * 16 + ncol;
      float v = acc[nt][r];
      if (bias) v += bias[nn];
      out[(size_t)mm * N + nn] = v;
    }
  }
}

// 1x1 conv (3->64) + bias + relu, fused with the identity pixel gather.
__global__ void k_conv(const float* __restrict__ img, const float* __restrict__ cw,
                       const float* __restrict__ cb, float* __restrict__ px) {
  int n = blockIdx.x * blockDim.x + threadIdx.x;
  if (n >= kN1) return;
  int b   = n / kHW;
  int rem = n - b * kHW;
  const float* ip = img + (size_t)b * 3 * kHW + rem;
  float i0 = ip[0], i1 = ip[kHW], i2 = ip[2 * kHW];
  float* op = px + (size_t)n * 64;
#pragma unroll
  for (int o = 0; o < 64; ++o) {
    float v = cb[o] + i0 * cw[o * 3] + i1 * cw[o * 3 + 1] + i2 * cw[o * 3 + 2];
    op[o] = fmaxf(v, 0.0f);
  }
}

__global__ void k_fill(float* __restrict__ p, float v, int n) {
  int i = blockIdx.x * blockDim.x + threadIdx.x;
  if (i < n) p[i] = v;
}

__global__ void k_deg(const int* __restrict__ dst, float* __restrict__ deg, int E) {
  int e = blockIdx.x * blockDim.x + threadIdx.x;
  if (e < E) atomAddF(&deg[dst[e]], 1.0f);
}

__global__ void k_dinv(float* __restrict__ deg, int n) {
  int i = blockIdx.x * blockDim.x + threadIdx.x;
  if (i < n) deg[i] = rsqrtf(fmaxf(deg[i], 1e-12f));
}

// self-loop term: agg[n,c] = x[n,c] * dinv[n]^2
__global__ void k_agg_self(const float* __restrict__ x, const float* __restrict__ dinv,
                           float* __restrict__ agg, int rows) {
  int idx = blockIdx.x * blockDim.x + threadIdx.x;
  if (idx >= rows * 128) return;
  int n = idx >> 7;
  float di = dinv[n];
  agg[idx] = x[idx] * di * di;
}

// edge scatter: agg[d,:] += x[s,:] * dinv[s]*dinv[d]; 1 thread = 1 edge x 4 channels
__global__ void k_agg_edge(const float* __restrict__ x, const float* __restrict__ dinv,
                           const int* __restrict__ src, const int* __restrict__ dst,
                           float* __restrict__ agg, int E) {
  int idx = blockIdx.x * blockDim.x + threadIdx.x;
  if (idx >= E * 32) return;
  int e = idx >> 5;
  int c = (idx & 31) * 4;
  int s = src[e], d = dst[e];
  float norm = dinv[s] * dinv[d];
  const float4 v = *reinterpret_cast<const float4*>(x + (size_t)s * 128 + c);
  float* ap = agg + (size_t)d * 128 + c;
  atomAddF(ap + 0, v.x * norm);
  atomAddF(ap + 1, v.y * norm);
  atomAddF(ap + 2, v.z * norm);
  atomAddF(ap + 3, v.w * norm);
}

// per-channel sum / sumsq (gcn bias cancels under BN, so stats are on agg directly)
__global__ void k_bn_stats(const float* __restrict__ a, float* __restrict__ stat, int rows) {
  int c = threadIdx.x;  // 128 threads = 128 channels
  float s = 0.0f, s2 = 0.0f;
  for (int r = blockIdx.x; r < rows; r += gridDim.x) {
    float v = a[(size_t)r * 128 + c];
    s += v;
    s2 += v * v;
  }
  atomAddF(&stat[c], s);
  atomAddF(&stat[128 + c], s2);
}

// h = h + relu( (agg - mean)*rsqrt(var+eps)*g + bb )   (in place)
__global__ void k_bn_apply(float* __restrict__ h, const float* __restrict__ agg,
                           const float* __restrict__ stat, const float* __restrict__ g,
                           const float* __restrict__ bb, int rows) {
  int idx = blockIdx.x * blockDim.x + threadIdx.x;
  if (idx >= rows * 128) return;
  int c = idx & 127;
  float inv_n = 1.0f / (float)rows;
  float mean = stat[c] * inv_n;
  float var  = stat[128 + c] * inv_n - mean * mean;
  float rstd = rsqrtf(var + 1e-5f);
  float t = (agg[idx] - mean) * rstd * g[c] + bb[c];
  h[idx] = h[idx] + fmaxf(t, 0.0f);
}

__global__ void k_pool_scatter(const float* __restrict__ h, const int* __restrict__ seg,
                               float* __restrict__ osum, float* __restrict__ ocnt, int rows) {
  int idx = blockIdx.x * blockDim.x + threadIdx.x;
  if (idx >= rows * 32) return;
  int n = idx >> 5;
  int c = (idx & 31) * 4;
  int sg = seg[n];
  const float4 v = *reinterpret_cast<const float4*>(h + (size_t)n * 128 + c);
  float* op = osum + (size_t)sg * 128 + c;
  atomAddF(op + 0, v.x);
  atomAddF(op + 1, v.y);
  atomAddF(op + 2, v.z);
  atomAddF(op + 3, v.w);
  if ((idx & 31) == 0) atomAddF(&ocnt[sg], 1.0f);
}

__global__ void k_pool_div(float* __restrict__ osum, const float* __restrict__ ocnt, int outn) {
  int idx = blockIdx.x * blockDim.x + threadIdx.x;
  if (idx >= outn * 128) return;
  osum[idx] = osum[idx] / fmaxf(ocnt[idx >> 7], 1.0f);
}

// logits[32,10] = hg2[32,128] @ W[128,10]  (N=10 not WMMA-tileable; trivial size)
__global__ void k_readout(const float* __restrict__ hg, const float* __restrict__ W,
                          float* __restrict__ out) {
  int t = blockIdx.x * blockDim.x + threadIdx.x;
  if (t >= kB * 10) return;
  int b = t / 10, o = t - b * 10;
  const float* hr = hg + (size_t)b * 128;
  float s = 0.0f;
  for (int c = 0; c < 128; ++c) s += hr[c] * W[c * 10 + o];
  out[t] = s;
}

extern "C" void kernel_launch(void* const* d_in, const int* in_sizes, int n_in,
                              void* d_out, int out_size, void* d_ws, size_t ws_size,
                              hipStream_t stream) {
  (void)in_sizes; (void)n_in; (void)out_size; (void)ws_size;
  const float* images   = (const float*)d_in[0];
  // d_in[1] pixel_data_where == identity decomposition (fused into k_conv)
  const int*   pei      = (const int*)d_in[2];   // [2, E1]
  const int*   pbatch   = (const int*)d_in[3];
  const int*   gei      = (const int*)d_in[4];   // [2, E2]
  const int*   gbatch   = (const int*)d_in[5];
  const float* conv_w   = (const float*)d_in[6];
  const float* conv_b   = (const float*)d_in[7];
  const float* emb1_W   = (const float*)d_in[8];
  const float* emb1_b   = (const float*)d_in[9];
  const float* gcn1_W   = (const float*)d_in[10];
  // d_in[11] gcn1_b cancels exactly inside BatchNorm
  const float* bn1_g    = (const float*)d_in[12];
  const float* bn1_b    = (const float*)d_in[13];
  const float* emb2_W   = (const float*)d_in[14];
  const float* emb2_b   = (const float*)d_in[15];
  const float* gcn2_W   = (const float*)d_in[16];
  // d_in[17] gcn2_b cancels exactly inside BatchNorm
  const float* bn2_g    = (const float*)d_in[18];
  const float* bn2_b    = (const float*)d_in[19];
  const float* readoutW = (const float*)d_in[20];

  char* wp = (char*)d_ws;
  auto take = [&](size_t bytes) {
    char* r = wp;
    wp += (bytes + 255) & ~(size_t)255;
    return (float*)r;
  };
  float* h     = take((size_t)kN1 * 128 * 4);
  float* xb    = take((size_t)kN1 * 128 * 4);   // also holds px[N1,64] for stage A
  float* aggb  = take((size_t)kN1 * 128 * 4);
  float* deg   = take((size_t)kN1 * 4);         // becomes dinv in place
  float* stat  = take(256 * 4);
  float* gpool = take((size_t)kG * 128 * 4);
  float* gh    = take((size_t)kG * 128 * 4);
  float* gx    = take((size_t)kG * 128 * 4);
  float* gagg  = take((size_t)kG * 128 * 4);
  float* gdeg  = take((size_t)kG * 4);
  float* gcnt  = take((size_t)kG * 4);
  float* bsum  = take((size_t)kB * 128 * 4);
  float* bcnt  = take((size_t)kB * 4);

  auto gcn_layer = [&](float* hh, float* xx, float* aa, float* dinv,
                       const int* src, const int* dst, int n, int E,
                       const float* W, const float* g, const float* bb) {
    k_gemm128<128><<<n / 128, 256, 0, stream>>>(hh, W, nullptr, xx, n);
    k_agg_self<<<(n * 128 + 255) / 256, 256, 0, stream>>>(xx, dinv, aa, n);
    k_agg_edge<<<(E * 32 + 255) / 256, 256, 0, stream>>>(xx, dinv, src, dst, aa, E);
    k_fill<<<1, 256, 0, stream>>>(stat, 0.0f, 256);
    k_bn_stats<<<256, 128, 0, stream>>>(aa, stat, n);
    k_bn_apply<<<(n * 128 + 255) / 256, 256, 0, stream>>>(hh, aa, stat, g, bb, n);
  };

  // Stage A: conv(3->64)+relu -> px, then h = px @ emb1_W + emb1_b
  k_conv<<<(kN1 + 255) / 256, 256, 0, stream>>>(images, conv_w, conv_b, xb);
  k_gemm128<64><<<kN1 / 128, 256, 0, stream>>>(xb, emb1_W, emb1_b, h, kN1);

  // pixel-graph degrees -> dinv
  k_fill<<<(kN1 + 255) / 256, 256, 0, stream>>>(deg, 1.0f, kN1);
  k_deg<<<(kE1 + 255) / 256, 256, 0, stream>>>(pei + kE1, deg, kE1);
  k_dinv<<<(kN1 + 255) / 256, 256, 0, stream>>>(deg, kN1);

  for (int i = 0; i < 2; ++i)
    gcn_layer(h, xb, aggb, deg, pei, pei + kE1, kN1, kE1,
              gcn1_W + (size_t)i * 128 * 128, bn1_g + i * 128, bn1_b + i * 128);

  // mean-pool pixels -> superpixels
  k_fill<<<(kG * 128 + 255) / 256, 256, 0, stream>>>(gpool, 0.0f, kG * 128);
  k_fill<<<(kG + 255) / 256, 256, 0, stream>>>(gcnt, 0.0f, kG);
  k_pool_scatter<<<(kN1 * 32 + 255) / 256, 256, 0, stream>>>(h, pbatch, gpool, gcnt, kN1);
  k_pool_div<<<(kG * 128 + 255) / 256, 256, 0, stream>>>(gpool, gcnt, kG);

  // h2 = gpool @ emb2_W + emb2_b
  k_gemm128<128><<<kG / 128, 256, 0, stream>>>(gpool, emb2_W, emb2_b, gh, kG);

  // graph-level degrees -> dinv
  k_fill<<<(kG + 255) / 256, 256, 0, stream>>>(gdeg, 1.0f, kG);
  k_deg<<<(kE2 + 255) / 256, 256, 0, stream>>>(gei + kE2, gdeg, kE2);
  k_dinv<<<(kG + 255) / 256, 256, 0, stream>>>(gdeg, kG);

  for (int i = 0; i < 2; ++i)
    gcn_layer(gh, gx, gagg, gdeg, gei, gei + kE2, kG, kE2,
              gcn2_W + (size_t)i * 128 * 128, bn2_g + i * 128, bn2_b + i * 128);

  // mean-pool superpixels -> batch, then readout
  k_fill<<<(kB * 128 + 255) / 256, 256, 0, stream>>>(bsum, 0.0f, kB * 128);
  k_fill<<<1, 256, 0, stream>>>(bcnt, 0.0f, kB);
  k_pool_scatter<<<(kG * 32 + 255) / 256, 256, 0, stream>>>(gh, gbatch, bsum, bcnt, kG);
  k_pool_div<<<(kB * 128 + 255) / 256, 256, 0, stream>>>(bsum, bcnt, kB);
  k_readout<<<2, 256, 0, stream>>>(bsum, readoutW, (float*)d_out);
}
